// ScepterVisionTransformer_85822036508852
// MI455X (gfx1250) — compile-verified, matrix operand order checked by reading, not run on hardware
//
#include <hip/hip_runtime.h>
#include <hip/hip_bf16.h>

typedef _Float16 h16_t;
typedef __attribute__((ext_vector_type(16))) _Float16 v16h;
typedef __attribute__((ext_vector_type(8)))  _Float16 v8h;
typedef __attribute__((ext_vector_type(8)))  float    v8f;

#define D_EMB   768
#define NHEAD   12
#define HD      64
#define T_PTS   6
#define NPATCH  392
#define SEQ     2352      // T_PTS * NPATCH
#define SEQ_PAD 2368      // SEQ padded to multiple of 64 (and 32)
#define BQ      2
#define MROWS   4704      // BQ * SEQ
#define MLP_HID 3072
#define FRAME_SZ 134456   // 56*49*49
#define RRELU_MLP  0.4f
#define RRELU_HEAD (11.0f/48.0f)

static __host__ int cdiv(int a, int b) { return (a + b - 1) / b; }

// ---------------------------------------------------------------------------
// Load one 16x32 f16 WMMA fragment (A layout; B uses the same per-lane shape
// with N in place of M). p must already include (row*ld + hi*8); elements
// 0..7 <- p[0..7], 8..15 <- p[16..23]: two aligned 16-byte loads.
// ---------------------------------------------------------------------------
__device__ __forceinline__ v16h load_frag(const h16_t* __restrict__ p)
{
    v8h lo = *(const v8h*)(p);
    v8h hh = *(const v8h*)(p + 16);
    v16h r;
    #pragma unroll
    for (int e = 0; e < 8; ++e) { r[e] = lo[e]; r[8 + e] = hh[e]; }
    return r;
}

// ---------------------------------------------------------------------------
// f16 WMMA GEMM: C[M,N] = act( A[M,K] @ B[N,K]^T + bias[N] + resid )
// One wave owns a 16x64 tile (4 accumulators), K stepped by 32.
// REQUIRES: M%16==0, N%64==0, K%32==0, lda/ldb%8==0 (16B-aligned loads).
// Branch-free inner loop: EXEC stays all-1s for every WMMA.
// ---------------------------------------------------------------------------
__global__ __launch_bounds__(256)
void gemm_f16_wmma(const h16_t* __restrict__ A, const h16_t* __restrict__ Bm,
                   const float* __restrict__ bias, const float* __restrict__ resid,
                   float* __restrict__ C, int M, int N, int K,
                   int lda, int ldb, int ldc, int act)
{
    int wid  = (blockIdx.x * blockDim.x + threadIdx.x) >> 5;
    int lane = threadIdx.x & 31;
    int ntn  = N >> 6;
    int totalW = (M >> 4) * ntn;
    if (wid >= totalW) return;                     // whole-wave exit
    int tm = (wid / ntn) << 4;
    int tn = (wid % ntn) << 6;
    int l  = lane & 15;
    int hi = lane >> 4;

    const h16_t* pA = A + (size_t)(tm + l) * lda + hi * 8;
    const h16_t* pB0 = Bm + (size_t)(tn + l) * ldb + hi * 8;
    const h16_t* pB1 = pB0 + (size_t)16 * ldb;
    const h16_t* pB2 = pB0 + (size_t)32 * ldb;
    const h16_t* pB3 = pB0 + (size_t)48 * ldb;

    v8f acc0 = v8f{0.f,0.f,0.f,0.f,0.f,0.f,0.f,0.f};
    v8f acc1 = acc0, acc2 = acc0, acc3 = acc0;

    for (int k0 = 0; k0 < K; k0 += 32) {
        v16h a  = load_frag(pA  + k0);
        v16h b0 = load_frag(pB0 + k0);
        v16h b1 = load_frag(pB1 + k0);
        v16h b2 = load_frag(pB2 + k0);
        v16h b3 = load_frag(pB3 + k0);
        acc0 = __builtin_amdgcn_wmma_f32_16x16x32_f16(false, a, false, b0, (short)0, acc0, false, false);
        acc1 = __builtin_amdgcn_wmma_f32_16x16x32_f16(false, a, false, b1, (short)0, acc1, false, false);
        acc2 = __builtin_amdgcn_wmma_f32_16x16x32_f16(false, a, false, b2, (short)0, acc2, false, false);
        acc3 = __builtin_amdgcn_wmma_f32_16x16x32_f16(false, a, false, b3, (short)0, acc3, false, false);
    }

    // Epilogue. C layout: VGPR r, lanes 0-15 -> row tm+r, col; lanes 16-31 -> row tm+r+8.
    v8f accs[4] = {acc0, acc1, acc2, acc3};
    #pragma unroll
    for (int j = 0; j < 4; ++j) {
        int col = tn + j * 16 + l;
        float bv = bias ? bias[col] : 0.f;
        #pragma unroll
        for (int r = 0; r < 8; ++r) {
            int row = tm + r + hi * 8;
            float v = accs[j][r] + bv;
            if (resid) v += resid[(size_t)row * ldc + col];
            if (act)   v = (v >= 0.f) ? v : RRELU_MLP * v;
            C[(size_t)row * ldc + col] = v;
        }
    }
}

// ---------------------------------------------------------------------------
__global__ void f32_to_f16(const float* __restrict__ in, h16_t* __restrict__ out, int n)
{
    for (int i = blockIdx.x * blockDim.x + threadIdx.x; i < n; i += gridDim.x * blockDim.x)
        out[i] = (h16_t)in[i];
}

// ---------------------------------------------------------------------------
// Patch embed: one block per token. Loads a 7x7x7 patch into LDS, each thread
// computes 3 of the 768 output channels; adds conv bias + pos_embed.
// ---------------------------------------------------------------------------
__global__ __launch_bounds__(256)
void patch_embed(const float* __restrict__ x, const float* __restrict__ w,
                 const float* __restrict__ wb, const float* __restrict__ pe,
                 float* __restrict__ tok)
{
    __shared__ float patch[343];
    int tk  = blockIdx.x;                 // 0..MROWS-1
    int b   = tk / SEQ;
    int seq = tk % SEQ;
    int t   = seq / NPATCH;
    int p   = seq % NPATCH;
    int gz = p / 49, gy = (p % 49) / 7, gx = p % 7;
    int f = b * T_PTS + t;
    const float* frame = x + (size_t)f * FRAME_SZ;
    for (int i = threadIdx.x; i < 343; i += 256) {
        int kz = i / 49, ky = (i % 49) / 7, kx = i % 7;
        patch[i] = frame[(gz * 7 + kz) * 2401 + (gy * 7 + ky) * 49 + (gx * 7 + kx)];
    }
    __syncthreads();
    for (int d = threadIdx.x; d < D_EMB; d += 256) {
        float s = wb[d];
        const float* wr = w + (size_t)d * 343;
        #pragma unroll 7
        for (int k = 0; k < 343; ++k) s += patch[k] * wr[k];
        tok[(size_t)tk * D_EMB + d] = s + pe[(size_t)seq * D_EMB + d];
    }
}

// ---------------------------------------------------------------------------
// LayerNorm over D=768, output f16 (GEMM A operand). One block per token.
// ---------------------------------------------------------------------------
__global__ __launch_bounds__(256)
void layernorm_f16(const float* __restrict__ X, const float* __restrict__ g,
                   const float* __restrict__ bt, h16_t* __restrict__ Y)
{
    __shared__ float red[256];
    int tid = threadIdx.x;
    int row = blockIdx.x;
    const float* xr = X + (size_t)row * D_EMB;
    float vals[3], s = 0.f, s2 = 0.f;
    #pragma unroll
    for (int i = 0; i < 3; ++i) {
        float v = xr[tid + i * 256];
        vals[i] = v; s += v; s2 += v * v;
    }
    red[tid] = s; __syncthreads();
    for (int o = 128; o > 0; o >>= 1) { if (tid < o) red[tid] += red[tid + o]; __syncthreads(); }
    float mean = red[0] / D_EMB; __syncthreads();
    red[tid] = s2; __syncthreads();
    for (int o = 128; o > 0; o >>= 1) { if (tid < o) red[tid] += red[tid + o]; __syncthreads(); }
    float var  = red[0] / D_EMB - mean * mean;
    float rstd = rsqrtf(var + 1e-6f);
    #pragma unroll
    for (int i = 0; i < 3; ++i) {
        int d = tid + i * 256;
        Y[(size_t)row * D_EMB + d] = (h16_t)((vals[i] - mean) * rstd * g[d] + bt[d]);
    }
}

// ---------------------------------------------------------------------------
// Final LayerNorm fused with dfc (768 -> 1). One block per token.
// ---------------------------------------------------------------------------
__global__ __launch_bounds__(256)
void final_ln_dfc(const float* __restrict__ X, const float* __restrict__ g,
                  const float* __restrict__ bt, const float* __restrict__ dw,
                  const float* __restrict__ db, float* __restrict__ zz)
{
    __shared__ float red[256];
    int tid = threadIdx.x;
    int row = blockIdx.x;
    const float* xr = X + (size_t)row * D_EMB;
    float vals[3], s = 0.f, s2 = 0.f;
    #pragma unroll
    for (int i = 0; i < 3; ++i) {
        float v = xr[tid + i * 256];
        vals[i] = v; s += v; s2 += v * v;
    }
    red[tid] = s; __syncthreads();
    for (int o = 128; o > 0; o >>= 1) { if (tid < o) red[tid] += red[tid + o]; __syncthreads(); }
    float mean = red[0] / D_EMB; __syncthreads();
    red[tid] = s2; __syncthreads();
    for (int o = 128; o > 0; o >>= 1) { if (tid < o) red[tid] += red[tid + o]; __syncthreads(); }
    float var  = red[0] / D_EMB - mean * mean;
    float rstd = rsqrtf(var + 1e-6f);
    __syncthreads();
    float part = 0.f;
    #pragma unroll
    for (int i = 0; i < 3; ++i) {
        int d = tid + i * 256;
        part += ((vals[i] - mean) * rstd * g[d] + bt[d]) * dw[d];
    }
    red[tid] = part; __syncthreads();
    for (int o = 128; o > 0; o >>= 1) { if (tid < o) red[tid] += red[tid + o]; __syncthreads(); }
    if (tid == 0) zz[row] = red[0] + db[0];
}

// ---------------------------------------------------------------------------
// Split qkv output (MROWS x 2304 f32) into:
//   q16  [bh][SEQ][HD]       (scaled by hd^-0.5)
//   k16  [bh][SEQ_PAD][HD]   (rows SEQ..SEQ_PAD-1 left unwritten; score GEMM
//                             computes junk there, softmax ignores those cols)
//   vT16 [bh][HD][SEQ_PAD]   (transposed; pad cols zeroed by vpad_zero)
// ---------------------------------------------------------------------------
__global__ void extract_qkv(const float* __restrict__ qkv, h16_t* __restrict__ q16,
                            h16_t* __restrict__ k16, h16_t* __restrict__ vT16)
{
    const float scale = 0.125f; // 64^-0.5
    int total = MROWS * 3 * D_EMB;
    for (int i = blockIdx.x * blockDim.x + threadIdx.x; i < total; i += gridDim.x * blockDim.x) {
        int m = i / (3 * D_EMB);
        int f = i % (3 * D_EMB);
        int s = f / D_EMB;
        int r = f % D_EMB;
        int h = r / HD, d = r % HD;
        int b = m / SEQ, seq = m % SEQ;
        int bh = b * NHEAD + h;
        float v = qkv[i];
        if (s == 0)      q16[((size_t)bh * SEQ + seq) * HD + d]     = (h16_t)(v * scale);
        else if (s == 1) k16[((size_t)bh * SEQ_PAD + seq) * HD + d] = (h16_t)v;
        else             vT16[((size_t)bh * HD + d) * SEQ_PAD + seq] = (h16_t)v;
    }
}

__global__ void vpad_zero(h16_t* __restrict__ vT16)
{
    int total = BQ * NHEAD * HD * (SEQ_PAD - SEQ);
    for (int i = blockIdx.x * blockDim.x + threadIdx.x; i < total; i += gridDim.x * blockDim.x) {
        int c  = SEQ + (i % (SEQ_PAD - SEQ));
        int d  = (i / (SEQ_PAD - SEQ)) % HD;
        int bh = i / ((SEQ_PAD - SEQ) * HD);
        vT16[((size_t)bh * HD + d) * SEQ_PAD + c] = (h16_t)0.f;
    }
}

// ---------------------------------------------------------------------------
// Row softmax of S (SEQ rows x SEQ_PAD stride f32, valid cols < SEQ)
// -> P16 (SEQ x SEQ_PAD f16, pad cols zeroed for the P@V GEMM's K padding).
// ---------------------------------------------------------------------------
__global__ __launch_bounds__(256)
void softmax_row(const float* __restrict__ S, h16_t* __restrict__ P)
{
    __shared__ float red[256];
    int tid = threadIdx.x;
    int row = blockIdx.x;
    const float* sr = S + (size_t)row * SEQ_PAD;
    float mx = -1e30f;
    for (int c = tid; c < SEQ; c += 256) mx = fmaxf(mx, sr[c]);
    red[tid] = mx; __syncthreads();
    for (int o = 128; o > 0; o >>= 1) { if (tid < o) red[tid] = fmaxf(red[tid], red[tid + o]); __syncthreads(); }
    mx = red[0]; __syncthreads();
    float sum = 0.f;
    for (int c = tid; c < SEQ; c += 256) sum += expf(sr[c] - mx);
    red[tid] = sum; __syncthreads();
    for (int o = 128; o > 0; o >>= 1) { if (tid < o) red[tid] += red[tid + o]; __syncthreads(); }
    float rinv = 1.f / red[0];
    for (int c = tid; c < SEQ_PAD; c += 256)
        P[(size_t)row * SEQ_PAD + c] = (c < SEQ) ? (h16_t)(expf(sr[c] - mx) * rinv) : (h16_t)0.f;
}

// ---------------------------------------------------------------------------
// z - mean_over_T + sinusoidal PE(T, NPATCH). One thread per (b,p).
// ---------------------------------------------------------------------------
__global__ void timemean_pe(const float* __restrict__ zz, float* __restrict__ zpe)
{
    int idx = blockIdx.x * blockDim.x + threadIdx.x;
    if (idx >= BQ * NPATCH) return;
    int b = idx / NPATCH, p = idx % NPATCH;
    float m = 0.f;
    #pragma unroll
    for (int t = 0; t < T_PTS; ++t) m += zz[((size_t)(b * T_PTS + t)) * NPATCH + p];
    m *= (1.f / T_PTS);
    int j = p >> 1;
    float div = expf(-(2.0f * j / (float)NPATCH) * logf(10000.0f));
    #pragma unroll
    for (int t = 0; t < T_PTS; ++t) {
        float pe = (p & 1) ? cosf(t * div) : sinf(t * div);
        size_t o = ((size_t)(b * T_PTS + t)) * NPATCH + p;
        zpe[o] = zz[o] - m + pe;
    }
}

// ---------------------------------------------------------------------------
// Depthwise ConvTranspose3d, gather form: out[o] = sum_{o = i*stride + j*dil} in[i]*w[j]
// Layout (b*T, D, H, W); w (T,1,k,k,k).
// ---------------------------------------------------------------------------
__global__ void convt_dw(const float* __restrict__ in, const float* __restrict__ w,
                         const float* __restrict__ bias, float* __restrict__ out,
                         int iD, int iH, int iW, int oD, int oH, int oW,
                         int k, int stride, int dil)
{
    int total = BQ * T_PTS * oD * oH * oW;
    int idx = blockIdx.x * blockDim.x + threadIdx.x;
    if (idx >= total) return;
    int x  = idx % oW;
    int y  = (idx / oW) % oH;
    int z  = (idx / (oW * oH)) % oD;
    int bc = idx / (oW * oH * oD);
    int c  = bc % T_PTS;
    float s = bias[c];
    const float* wc = w + (size_t)c * k * k * k;
    for (int kz = 0; kz < k; ++kz) {
        int tz = z - kz * dil;
        if (tz < 0 || (tz % stride)) continue;
        int iz = tz / stride; if (iz >= iD) continue;
        for (int ky = 0; ky < k; ++ky) {
            int ty = y - ky * dil;
            if (ty < 0 || (ty % stride)) continue;
            int iy = ty / stride; if (iy >= iH) continue;
            for (int kx = 0; kx < k; ++kx) {
                int tx = x - kx * dil;
                if (tx < 0 || (tx % stride)) continue;
                int ix = tx / stride; if (ix >= iW) continue;
                s += in[((size_t)bc * iD + iz) * iH * iW + iy * iW + ix]
                   * wc[(kz * k + ky) * k + kx];
            }
        }
    }
    out[idx] = s;
}

// ---------------------------------------------------------------------------
// Nearest resize (47,45,45)->(56,49,49), per-channel affine, RReLU head,
// and transpose to (b,1,56,49,49,T).
// ---------------------------------------------------------------------------
__global__ void head_out(const float* __restrict__ u3, const float* __restrict__ hw,
                         const float* __restrict__ hb, float* __restrict__ out)
{
    int total = BQ * 56 * 49 * 49;
    int idx = blockIdx.x * blockDim.x + threadIdx.x;
    if (idx >= total) return;
    int x = idx % 49;
    int y = (idx / 49) % 49;
    int z = (idx / 2401) % 56;
    int b = idx / (2401 * 56);
    int sz = (z * 47) / 56;
    int sy = (y * 45) / 49;
    int sx = (x * 45) / 49;
    #pragma unroll
    for (int t = 0; t < T_PTS; ++t) {
        float v = u3[((size_t)(b * T_PTS + t) * 47 + sz) * 2025 + sy * 45 + sx];
        v = v * hw[t] + hb[t];
        v = (v >= 0.f) ? v : RRELU_HEAD * v;
        out[(size_t)idx * T_PTS + t] = v;
    }
}

// ---------------------------------------------------------------------------
extern "C" void kernel_launch(void* const* d_in, const int* in_sizes, int n_in,
                              void* d_out, int out_size, void* d_ws, size_t ws_size,
                              hipStream_t stream)
{
    (void)in_sizes; (void)n_in; (void)out_size; (void)ws_size;
    const float* x        = (const float*)d_in[0];
    const float* pconv_w  = (const float*)d_in[1];
    const float* pconv_b  = (const float*)d_in[2];
    const float* pos_emb  = (const float*)d_in[3];
    const float* ln1_g    = (const float*)d_in[4];
    const float* ln1_b    = (const float*)d_in[5];
    const float* qkv_w    = (const float*)d_in[6];
    const float* qkv_b    = (const float*)d_in[7];
    const float* proj_w   = (const float*)d_in[8];
    const float* proj_b   = (const float*)d_in[9];
    const float* ln2_g    = (const float*)d_in[10];
    const float* ln2_b    = (const float*)d_in[11];
    const float* fc1_w    = (const float*)d_in[12];
    const float* fc1_b    = (const float*)d_in[13];
    const float* fc2_w    = (const float*)d_in[14];
    const float* fc2_b    = (const float*)d_in[15];
    const float* normf_g  = (const float*)d_in[16];
    const float* normf_b  = (const float*)d_in[17];
    const float* dfc_w    = (const float*)d_in[18];
    const float* dfc_b    = (const float*)d_in[19];
    const float* up1_w    = (const float*)d_in[20];
    const float* up1_b    = (const float*)d_in[21];
    const float* up2_w    = (const float*)d_in[22];
    const float* up2_b    = (const float*)d_in[23];
    const float* up3_w    = (const float*)d_in[24];
    const float* up3_b    = (const float*)d_in[25];
    const float* hconv_w  = (const float*)d_in[26];
    const float* hconv_b  = (const float*)d_in[27];
    float* outp = (float*)d_out;

    // ---- workspace arena ----
    size_t off = 0;
    char* base = (char*)d_ws;
    auto take = [&](size_t bytes) -> void* {
        void* p = base + off;
        off += (bytes + 255) & ~(size_t)255;
        return p;
    };
    float* tok  = (float*)take((size_t)MROWS * D_EMB * 4);
    h16_t* a16  = (h16_t*)take((size_t)MROWS * MLP_HID * 2);
    h16_t* w16  = (h16_t*)take((size_t)MLP_HID * D_EMB * 2);
    float* big  = (float*)take((size_t)MROWS * MLP_HID * 4);   // qkv out / mlp hidden
    h16_t* q16  = (h16_t*)take((size_t)BQ * NHEAD * SEQ * HD * 2);
    h16_t* k16  = (h16_t*)take((size_t)BQ * NHEAD * SEQ_PAD * HD * 2);
    h16_t* vT16 = (h16_t*)take((size_t)BQ * NHEAD * HD * SEQ_PAD * 2);
    float* Sbuf = (float*)take((size_t)SEQ * SEQ_PAD * 4);
    h16_t* P16  = (h16_t*)take((size_t)SEQ * SEQ_PAD * 2);
    float* obuf = (float*)take((size_t)MROWS * D_EMB * 4);
    float* zz   = (float*)take((size_t)MROWS * 4);
    float* zpe  = (float*)take((size_t)MROWS * 4);
    float* u1   = (float*)take((size_t)12 * 14 * 13 * 13 * 4);
    float* u2   = (float*)take((size_t)12 * 31 * 29 * 29 * 4);
    float* u3   = (float*)take((size_t)12 * 47 * 45 * 45 * 4);

    auto gemm_blocks = [](int M, int N) {
        int waves = (M / 16) * (N / 64);
        return cdiv(waves, 8);
    };

    // ---- patch embed + pos embed ----
    patch_embed<<<MROWS, 256, 0, stream>>>(x, pconv_w, pconv_b, pos_emb, tok);

    // ---- transformer layers ----
    for (int L = 0; L < 2; ++L) {
        // LN1 -> f16
        layernorm_f16<<<MROWS, 256, 0, stream>>>(tok, ln1_g + L * D_EMB, ln1_b + L * D_EMB, a16);
        // QKV
        f32_to_f16<<<cdiv(3 * D_EMB * D_EMB, 256), 256, 0, stream>>>(qkv_w + (size_t)L * 3 * D_EMB * D_EMB, w16, 3 * D_EMB * D_EMB);
        gemm_f16_wmma<<<gemm_blocks(MROWS, 3 * D_EMB), 256, 0, stream>>>(
            a16, w16, qkv_b + L * 3 * D_EMB, nullptr, big,
            MROWS, 3 * D_EMB, D_EMB, D_EMB, D_EMB, 3 * D_EMB, 0);
        extract_qkv<<<cdiv(MROWS * 3 * D_EMB, 256), 256, 0, stream>>>(big, q16, k16, vT16);
        vpad_zero<<<cdiv(BQ * NHEAD * HD * (SEQ_PAD - SEQ), 256), 256, 0, stream>>>(vT16);

        // attention per (b,h): S = q@k^T (scale folded into q), softmax, O = P@V
        for (int bh = 0; bh < BQ * NHEAD; ++bh) {
            const h16_t* qh = q16 + (size_t)bh * SEQ * HD;
            const h16_t* kh = k16 + (size_t)bh * SEQ_PAD * HD;
            const h16_t* vh = vT16 + (size_t)bh * HD * SEQ_PAD;
            // N padded to SEQ_PAD: cols SEQ..SEQ_PAD-1 are junk, softmax skips them
            gemm_f16_wmma<<<gemm_blocks(SEQ, SEQ_PAD), 256, 0, stream>>>(
                qh, kh, nullptr, nullptr, Sbuf, SEQ, SEQ_PAD, HD, HD, HD, SEQ_PAD, 0);
            softmax_row<<<SEQ, 256, 0, stream>>>(Sbuf, P16);
            int b = bh / NHEAD, h = bh % NHEAD;
            gemm_f16_wmma<<<gemm_blocks(SEQ, HD), 256, 0, stream>>>(
                P16, vh, nullptr, nullptr,
                obuf + (size_t)b * SEQ * D_EMB + h * HD,
                SEQ, HD, SEQ_PAD, SEQ_PAD, SEQ_PAD, D_EMB, 0);
        }

        // proj + residual
        f32_to_f16<<<cdiv(MROWS * D_EMB, 256), 256, 0, stream>>>(obuf, a16, MROWS * D_EMB);
        f32_to_f16<<<cdiv(D_EMB * D_EMB, 256), 256, 0, stream>>>(proj_w + (size_t)L * D_EMB * D_EMB, w16, D_EMB * D_EMB);
        gemm_f16_wmma<<<gemm_blocks(MROWS, D_EMB), 256, 0, stream>>>(
            a16, w16, proj_b + L * D_EMB, tok, tok,
            MROWS, D_EMB, D_EMB, D_EMB, D_EMB, D_EMB, 0);

        // LN2 -> MLP
        layernorm_f16<<<MROWS, 256, 0, stream>>>(tok, ln2_g + L * D_EMB, ln2_b + L * D_EMB, a16);
        f32_to_f16<<<cdiv(MLP_HID * D_EMB, 256), 256, 0, stream>>>(fc1_w + (size_t)L * MLP_HID * D_EMB, w16, MLP_HID * D_EMB);
        gemm_f16_wmma<<<gemm_blocks(MROWS, MLP_HID), 256, 0, stream>>>(
            a16, w16, fc1_b + L * MLP_HID, nullptr, big,
            MROWS, MLP_HID, D_EMB, D_EMB, D_EMB, MLP_HID, 1);  // fused RReLU(0.4)
        f32_to_f16<<<cdiv(MROWS * MLP_HID, 256), 256, 0, stream>>>(big, a16, MROWS * MLP_HID);
        f32_to_f16<<<cdiv(D_EMB * MLP_HID, 256), 256, 0, stream>>>(fc2_w + (size_t)L * D_EMB * MLP_HID, w16, D_EMB * MLP_HID);
        gemm_f16_wmma<<<gemm_blocks(MROWS, D_EMB), 256, 0, stream>>>(
            a16, w16, fc2_b + L * D_EMB, tok, tok,
            MROWS, D_EMB, MLP_HID, MLP_HID, MLP_HID, D_EMB, 0);
    }

    // ---- decoder head ----
    final_ln_dfc<<<MROWS, 256, 0, stream>>>(tok, normf_g, normf_b, dfc_w, dfc_b, zz);
    timemean_pe<<<cdiv(BQ * NPATCH, 256), 256, 0, stream>>>(zz, zpe);
    convt_dw<<<cdiv(12 * 14 * 13 * 13, 256), 256, 0, stream>>>(
        zpe, up1_w, up1_b, u1, 8, 7, 7, 14, 13, 13, 7, 1, 1);
    convt_dw<<<cdiv(12 * 31 * 29 * 29, 256), 256, 0, stream>>>(
        u1, up2_w, up2_b, u2, 14, 13, 13, 31, 29, 29, 5, 2, 1);
    convt_dw<<<cdiv(12 * 47 * 45 * 45, 256), 256, 0, stream>>>(
        u2, up3_w, up3_b, u3, 31, 29, 29, 47, 45, 45, 9, 1, 2);
    head_out<<<cdiv(BQ * 56 * 49 * 49, 256), 256, 0, stream>>>(u3, hconv_w, hconv_b, outp);
}
